// LinearAttention_9019431322131
// MI455X (gfx1250) — matrix operand order repaired and checked by reading, not compiled, not run on hardware
//
#include <hip/hip_runtime.h>
#include <hip/hip_bf16.h>

// ---------------------------------------------------------------------------
// Linear attention (B=4, S=4096, D=1024, H=16, Dh=64) for gfx1250 (MI455X).
// All GEMMs on v_wmma_f32_16x16x32_bf16 (f32 accumulate).
//  - weights pre-swizzled into B-fragment layout: inner loops are pure
//    global_load_b128 + WMMA (verified in round-2 codegen)
//  - k/v/kv stored transposed so every fragment load/store is contiguous
//  - kv reduction split 8-way over S for occupancy (512 WGs), deterministic
//    f32 partials + serial 8-way sum (no atomics)
//  - K3 stages the per-head 8KB kv tile in LDS via global_load_async_to_lds
//    (ASYNCcnt) and reads B-fragments with ds_load_b128
// ---------------------------------------------------------------------------

typedef __attribute__((ext_vector_type(16))) __bf16 v16bf;
typedef __attribute__((ext_vector_type(8)))  __bf16 v8bf;
typedef __attribute__((ext_vector_type(8)))  float  v8f;
typedef __attribute__((ext_vector_type(4)))  float  f4;

#define WMMA_BF16(A, Bm, C) \
  __builtin_amdgcn_wmma_f32_16x16x32_bf16(false, (A), false, (Bm), (short)0, (C), false, false)

namespace {
constexpr int BATCH = 4;
constexpr int SEQ   = 4096;
constexpr int DIM   = 1024;
constexpr int HEADS = 16;
constexpr int DH    = 64;
constexpr int INNER = HEADS * DH;   // 1024
constexpr int N3    = 3 * INNER;    // 3072
constexpr int MROWS = BATCH * SEQ;  // 16384
constexpr int KVSPLIT = 8;          // S-chunks for the kv reduction
constexpr int KVCH    = SEQ / KVSPLIT;  // 512
}

// A fragment from a row-major bf16 row pointer (already offset to k0).
// A 16x32 layout (contiguous k order, matching B): half=0 -> K{0..15}, half=1 -> K{16..31}:
// two contiguous 16B runs per lane.
static __device__ __forceinline__ v16bf load_a_row(const __bf16* __restrict__ p,
                                                   int half) {
  const v8bf lo = *(const v8bf*)(p + 16 * half);
  const v8bf hi = *(const v8bf*)(p + 16 * half + 8);
  return __builtin_shufflevector(lo, hi, 0, 1, 2, 3, 4, 5, 6, 7,
                                 8, 9, 10, 11, 12, 13, 14, 15);
}

// Packed-B layout: frag for (kt, nt) stored as [lane][e] contiguous 32B.
static __device__ __forceinline__ v16bf load_b_packed(const __bf16* __restrict__ wp,
                                                      int kt, int nt, int nTiles,
                                                      int lane) {
  return *(const v16bf*)(wp + (((size_t)kt * nTiles + nt) * 32 + lane) * 16);
}

static __device__ __forceinline__ float elu1(float v) {
  return (v > 0.0f) ? (v + 1.0f) : __expf(v);
}

// ---------------------------------------------------------------------------
// P1: x f32 -> bf16 row-major (one-time; 8 elems/thread)
// ---------------------------------------------------------------------------
__global__ __launch_bounds__(256) void p_cvt_x(const float* __restrict__ x,
                                               __bf16* __restrict__ xb) {
  const size_t i = ((size_t)blockIdx.x * 256 + threadIdx.x) * 8;
  const f4 a = *(const f4*)(x + i);
  const f4 b = *(const f4*)(x + i + 4);
  v8bf o;
#pragma unroll
  for (int j = 0; j < 4; ++j) { o[j] = (__bf16)a[j]; o[4 + j] = (__bf16)b[j]; }
  *(v8bf*)(xb + i) = o;
}

// ---------------------------------------------------------------------------
// P2: pack weight f32 [K,N] into B-fragment-ordered bf16 (one thread per
// (ktile, ntile, lane), writes its contiguous 16-element fragment slice).
// ---------------------------------------------------------------------------
__global__ __launch_bounds__(256) void p_pack_w(const float* __restrict__ w,
                                                __bf16* __restrict__ dst,
                                                int N, int nTiles) {
  const int t    = blockIdx.x * 256 + threadIdx.x;
  const int lane = t & 31;
  const int tile = t >> 5;              // kt * nTiles + nt
  const int nt   = tile % nTiles;
  const int kt   = tile / nTiles;
  const int half = lane >> 4;
  const int ln   = lane & 15;
  const int col  = nt * 16 + ln;
  const int kb   = kt * 32 + 16 * half;
  v16bf o;
#pragma unroll
  for (int e = 0; e < 16; ++e) o[e] = (__bf16)w[(size_t)(kb + e) * N + col];
  *(v16bf*)(dst + ((size_t)tile * 32 + lane) * 16) = o;
}

// ---------------------------------------------------------------------------
// K1: qkv = x @ w_qkv (M=16384,K=1024,N=3072), fused elu+1.
//  q -> [B,H,S,Dh]; k,v -> [B,H,Dh,S] transposed (contiguous b128 stores)
// ---------------------------------------------------------------------------
__global__ __launch_bounds__(128) void k1_qkv(const __bf16* __restrict__ xb,
                                              const __bf16* __restrict__ wp,
                                              __bf16* __restrict__ qws,
                                              __bf16* __restrict__ kT,
                                              __bf16* __restrict__ vT) {
  const int wave = threadIdx.x >> 5;
  const int lane = threadIdx.x & 31;
  const int half = lane >> 4;
  const int ln   = lane & 15;
  const int mBase = blockIdx.x * 64 + wave * 16;

  const __bf16* arow = xb + (size_t)(mBase + ln) * DIM;
  v8f acc[4] = {};
  for (int kt = 0; kt < DIM / 32; ++kt) {
    __builtin_prefetch(arow + kt * 32 + 256, 0, 1);  // global_prefetch_b8
    const v16bf a = load_a_row(arow + kt * 32, half);
#pragma unroll
    for (int j = 0; j < 4; ++j) {
      const v16bf b = load_b_packed(wp, kt, blockIdx.y * 4 + j, N3 / 16, lane);
      acc[j] = WMMA_BF16(a, b, acc[j]);
    }
  }

#pragma unroll
  for (int j = 0; j < 4; ++j) {
    const int col = blockIdx.y * 64 + j * 16 + ln;
    const int sec = col >> 10;          // 0=q, 1=k, 2=v
    const int c   = col & (INNER - 1);
    const int h   = c >> 6;
    const int dh  = c & (DH - 1);
    const int m0  = mBase + 8 * half;   // 8 consecutive rows per lane
    const int b   = m0 >> 12;
    const int s0  = m0 & (SEQ - 1);
    const size_t bh = (size_t)b * HEADS + h;
    if (sec == 0) {
      __bf16* dst = qws + (bh * SEQ + s0) * DH + dh;
#pragma unroll
      for (int r = 0; r < 8; ++r)
        dst[(size_t)r * DH] = (__bf16)elu1(acc[j][r]);
    } else {
      __bf16* dst = ((sec == 1) ? kT : vT) + (bh * DH + dh) * SEQ + s0;
      v8bf o;
#pragma unroll
      for (int r = 0; r < 8; ++r) {
        float v = acc[j][r];
        if (sec == 1) v = elu1(v);
        o[r] = (__bf16)v;
      }
      *(v8bf*)dst = o;   // contiguous along S
    }
  }
}

// ---------------------------------------------------------------------------
// K2: partial kv over one S-chunk: part[bh][sp][e][d] (f32, deterministic).
// Grid (B*H, KVSPLIT), block 128.
// ---------------------------------------------------------------------------
__global__ __launch_bounds__(128) void k2_kv(const __bf16* __restrict__ kT,
                                             const __bf16* __restrict__ vT,
                                             float* __restrict__ part) {
  const int bh   = blockIdx.x;
  const int sp   = blockIdx.y;
  const int wave = threadIdx.x >> 5;
  const int lane = threadIdx.x & 31;
  const int half = lane >> 4;
  const int ln   = lane & 15;
  const int dRow = wave * 16 + ln;

  const __bf16* arow = kT + ((size_t)bh * DH + dRow) * SEQ;
  v8f acc[4] = {};
  const int sBeg = sp * KVCH;
  for (int s0 = sBeg; s0 < sBeg + KVCH; s0 += 32) {
    const v16bf a = load_a_row(arow + s0, half);
#pragma unroll
    for (int j = 0; j < 4; ++j) {
      const __bf16* brow = vT + ((size_t)bh * DH + j * 16 + ln) * SEQ;
      const v16bf b = *(const v16bf*)(brow + s0 + 16 * half);
      acc[j] = WMMA_BF16(a, b, acc[j]);
    }
  }

#pragma unroll
  for (int j = 0; j < 4; ++j) {
    const int e  = j * 16 + ln;          // column (v-feature)
    const int d0 = wave * 16 + 8 * half; // 8 consecutive d per lane
    float* dst = part + (((size_t)bh * KVSPLIT + sp) * DH + e) * DH + d0;
    *(v8f*)dst = acc[j];                 // 32B contiguous store
  }
}

// ---------------------------------------------------------------------------
// K2c: kvT[bh][e][d] = bf16( sum_sp part[bh][sp][e][d] )
// ---------------------------------------------------------------------------
__global__ __launch_bounds__(256) void k2c_reduce(const float* __restrict__ part,
                                                  __bf16* __restrict__ kvT) {
  const int i  = blockIdx.x * 256 + threadIdx.x;   // 0 .. 64*4096-1
  const int bh = i >> 12;
  const int ed = i & 4095;
  float s = 0.0f;
#pragma unroll
  for (int sp = 0; sp < KVSPLIT; ++sp)
    s += part[((size_t)bh * KVSPLIT + sp) * (DH * DH) + ed];
  kvT[(size_t)bh * (DH * DH) + ed] = (__bf16)s;
}

// ---------------------------------------------------------------------------
// K2b: z[bh][d] = sum_s kT[bh][d][s]; one block per (bh,d), LDS reduction.
// ---------------------------------------------------------------------------
__global__ __launch_bounds__(64) void k2b_z(const __bf16* __restrict__ kT,
                                            float* __restrict__ z) {
  const int bh = blockIdx.x >> 6;
  const int d  = blockIdx.x & (DH - 1);
  const int tid = threadIdx.x;
  const __bf16* row = kT + ((size_t)bh * DH + d) * SEQ;
  float acc = 0.0f;
  for (int s = tid; s < SEQ; s += 64) acc += (float)row[s];
  __shared__ float red[64];
  red[tid] = acc;
  __syncthreads();
#pragma unroll
  for (int off = 32; off > 0; off >>= 1) {
    if (tid < off) red[tid] += red[tid + off];
    __syncthreads();
  }
  if (tid == 0) z[bh * DH + d] = red[0];
}

// ---------------------------------------------------------------------------
// K3: num = q @ kv; den = q.z + 1e-6; attn = num/den -> [B,S,H*Dh] bf16.
// kv tile (8KB) staged in LDS via global_load_async_to_lds_b128 (ASYNCcnt),
// B-fragments then come from ds_load_b128.
// ---------------------------------------------------------------------------
__global__ __launch_bounds__(128) void k3_attn(const __bf16* __restrict__ qws,
                                               const __bf16* __restrict__ kvT,
                                               const float* __restrict__ z,
                                               __bf16* __restrict__ attn) {
  const int bh   = blockIdx.x;
  const int srow = blockIdx.y * 64;
  const int wave = threadIdx.x >> 5;
  const int lane = threadIdx.x & 31;
  const int half = lane >> 4;
  const int ln   = lane & 15;
  const int b = bh >> 4;
  const int h = bh & (HEADS - 1);
  const __bf16* qb = qws + (size_t)bh * SEQ * DH;
  const float*  zb = z + bh * DH;

  __shared__ __align__(16) __bf16 kvLds[DH * DH];   // 8 KB
  __shared__ float denomLds[64];

  // 1) kick off the async DMA of this head's kv tile into LDS
  {
    const __bf16* src = kvT + (size_t)bh * (DH * DH);
    const unsigned ldsBase = (unsigned)(size_t)(&kvLds[0]);
#pragma unroll
    for (int i = 0; i < 4; ++i) {
      const int eoff = (threadIdx.x + i * 128) * 8;          // 8 bf16 = 16B
      const unsigned long long g = (unsigned long long)(size_t)(src + eoff);
      asm volatile("global_load_async_to_lds_b128 %0, %1, off"
                   :: "v"(ldsBase + eoff * 2), "v"(g) : "memory");
    }
  }

  // 2) overlap: denominators for this block's 64 rows
  if (threadIdx.x < 64) {
    const int s = srow + threadIdx.x;
    float acc = 1e-6f;
    for (int d = 0; d < DH; ++d) acc += (float)qb[(size_t)s * DH + d] * zb[d];
    denomLds[threadIdx.x] = acc;
  }

  asm volatile("s_wait_asynccnt 0x0" ::: "memory");
  __syncthreads();

  const int mBase = srow + wave * 16;
  const __bf16* arow = qb + (size_t)(mBase + ln) * DH;
  v8f acc[4] = {};
#pragma unroll
  for (int k0 = 0; k0 < DH; k0 += 32) {
    const v16bf a = load_a_row(arow + k0, half);
#pragma unroll
    for (int j = 0; j < 4; ++j) {
      const v16bf bm = *(const v16bf*)(kvLds + (j * 16 + ln) * DH
                                       + k0 + 16 * half);   // ds_load_b128 x2
      acc[j] = WMMA_BF16(a, bm, acc[j]);
    }
  }

#pragma unroll
  for (int j = 0; j < 4; ++j) {
    const int col = j * 16 + ln;
#pragma unroll
    for (int r = 0; r < 8; ++r) {
      const int sl = wave * 16 + 8 * half + r;
      const int s  = srow + sl;
      const float v = acc[j][r] / denomLds[sl];
      attn[((size_t)(b * SEQ + s)) * INNER + h * DH + col] = (__bf16)v;
    }
  }
}

// ---------------------------------------------------------------------------
// K4: out = attn @ w_out + b_out  (M=16384, K=1024, N=1024, f32 out)
// ---------------------------------------------------------------------------
__global__ __launch_bounds__(128) void k4_out(const __bf16* __restrict__ attn,
                                              const __bf16* __restrict__ wp,
                                              const float* __restrict__ bias,
                                              float* __restrict__ out) {
  const int wave = threadIdx.x >> 5;
  const int lane = threadIdx.x & 31;
  const int half = lane >> 4;
  const int ln   = lane & 15;
  const int mBase = blockIdx.x * 64 + wave * 16;
  const int nBase = blockIdx.y * 64;

  const __bf16* arow = attn + (size_t)(mBase + ln) * INNER;
  v8f acc[4] = {};
  for (int kt = 0; kt < INNER / 32; ++kt) {
    __builtin_prefetch(arow + kt * 32 + 256, 0, 1);
    const v16bf a = load_a_row(arow + kt * 32, half);
#pragma unroll
    for (int j = 0; j < 4; ++j) {
      const v16bf b = load_b_packed(wp, kt, blockIdx.y * 4 + j, DIM / 16, lane);
      acc[j] = WMMA_BF16(a, b, acc[j]);
    }
  }

#pragma unroll
  for (int j = 0; j < 4; ++j) {
    const int col = nBase + j * 16 + ln;
    const float bv = bias[col];
#pragma unroll
    for (int r = 0; r < 8; ++r) {
      const int m = mBase + 8 * half + r;
      out[(size_t)m * DIM + col] = acc[j][r] + bv;
    }
  }
}

// ---------------------------------------------------------------------------
extern "C" void kernel_launch(void* const* d_in, const int* in_sizes, int n_in,
                              void* d_out, int out_size, void* d_ws, size_t ws_size,
                              hipStream_t stream) {
  (void)in_sizes; (void)n_in; (void)out_size; (void)ws_size;
  const float* x     = (const float*)d_in[0];
  const float* w_qkv = (const float*)d_in[1];
  const float* w_out = (const float*)d_in[2];
  const float* b_out = (const float*)d_in[3];
  float* out = (float*)d_out;

  char* ws = (char*)d_ws;
  const size_t MB = 1ull << 20;
  __bf16* xbf    = (__bf16*)(ws);              // 32 MiB (reused as attn)
  __bf16* attn   = xbf;                        // alias: x dead after K1
  __bf16* wqkvp  = (__bf16*)(ws + 32 * MB);    // 6 MiB packed
  __bf16* woutp  = (__bf16*)(ws + 38 * MB);    // 2 MiB packed
  __bf16* qws    = (__bf16*)(ws + 40 * MB);    // 32 MiB [B,H,S,Dh]
  __bf16* kT     = (__bf16*)(ws + 72 * MB);    // 32 MiB [B,H,Dh,S]
  __bf16* vT     = (__bf16*)(ws + 104 * MB);   // 32 MiB [B,H,Dh,S]
  __bf16* kvT    = (__bf16*)(ws + 136 * MB);   // 0.5 MiB
  float*  zws    = (float*)(ws + 137 * MB);    // 16 KiB
  float*  kvpart = (float*)(ws + 138 * MB);    // 8 MiB f32 partials

  p_cvt_x<<<dim3(MROWS * DIM / (256 * 8)), dim3(256), 0, stream>>>(x, xbf);
  p_pack_w<<<dim3((DIM / 32) * (N3 / 16) * 32 / 256), dim3(256), 0, stream>>>(
      w_qkv, wqkvp, N3, N3 / 16);
  p_pack_w<<<dim3((INNER / 32) * (DIM / 16) * 32 / 256), dim3(256), 0, stream>>>(
      w_out, woutp, DIM, DIM / 16);

  dim3 blk(128);
  k1_qkv<<<dim3(MROWS / 64, N3 / 64), blk, 0, stream>>>(xbf, wqkvp, qws, kT, vT);
  k2_kv<<<dim3(BATCH * HEADS, KVSPLIT), blk, 0, stream>>>(kT, vT, kvpart);
  k2c_reduce<<<dim3(BATCH * HEADS * DH * DH / 256), dim3(256), 0, stream>>>(
      kvpart, kvT);
  k2b_z<<<dim3(BATCH * HEADS * DH), dim3(64), 0, stream>>>(kT, zws);
  k3_attn<<<dim3(BATCH * HEADS, SEQ / 64), blk, 0, stream>>>(qws, kvT, zws, attn);
  k4_out<<<dim3(MROWS / 64, DIM / 64), blk, 0, stream>>>(attn, woutp, b_out, out);
}
